// OrthoLinear_1099511628339
// MI455X (gfx1250) — matrix operand order repaired and checked by reading, not compiled, not run on hardware
//
#include <hip/hip_runtime.h>

typedef __attribute__((ext_vector_type(16))) _Float16 v16h;
typedef __attribute__((ext_vector_type(8)))  _Float16 v8h;
typedef __attribute__((ext_vector_type(8)))  float    v8f;

#define O_FEAT 4096
#define I_FEAT 4096
#define LDSTR  40   // 32 halves payload + 8 halves pad (80B row stride, 16B aligned)

// ---------------------------------------------------------------------------
// Kernel 1: dequantize packed int4 -> f16 W[O][I]. One thread per 4 packed
// bytes => 8 output halves (one 16B store).
// ---------------------------------------------------------------------------
__global__ __launch_bounds__(256) void ortho_dequant_kernel(
    const unsigned char* __restrict__ packed,
    const float* __restrict__ scales,
    _Float16* __restrict__ W) {
  const int idx = blockIdx.x * blockDim.x + threadIdx.x;      // 0 .. O*I/8-1
  const unsigned int v = ((const unsigned int*)packed)[idx];
  const int o  = (idx * 4) >> 11;        // / (I_FEAT/2)
  const int jb = (idx * 4) & 2047;       // packed-byte index within row
  const float s = scales[o];
  v8h out;
#pragma unroll
  for (int b = 0; b < 4; ++b) {
    const unsigned int byt = (v >> (8 * b)) & 0xFFu;
    out[2 * b]     = (_Float16)(((int)(byt & 0xFu) - 8) * s);
    out[2 * b + 1] = (_Float16)(((int)(byt >> 4)   - 8) * s);
  }
  *(v8h*)(W + (size_t)o * I_FEAT + (size_t)jb * 2) = out;
}

// ---------------------------------------------------------------------------
// Kernel 2: scatter-add alpha * COO residual into W using the CDNA5 packed
// f16 global atomic (duplicate (row,col) entries must accumulate).
// ---------------------------------------------------------------------------
__global__ __launch_bounds__(256) void ortho_scatter_kernel(
    const _Float16* __restrict__ vals,
    const int* __restrict__ rows,
    const int* __restrict__ cols,
    const float* __restrict__ alpha,
    _Float16* __restrict__ W, int nnz) {
  const int i = blockIdx.x * blockDim.x + threadIdx.x;
  if (i >= nnz) return;
  const float fv = (float)vals[i] * alpha[0];
  const int r = rows[i];
  const int c = cols[i];
  const _Float16 hv = (_Float16)fv;
  const unsigned short bits = __builtin_bit_cast(unsigned short, hv);
  const unsigned int pk = (c & 1) ? ((unsigned int)bits << 16) : (unsigned int)bits;
  const unsigned long long addr =
      (unsigned long long)(W + (size_t)r * I_FEAT + (size_t)(c & ~1));
  asm volatile("global_atomic_pk_add_f16 %0, %1, off"
               :: "v"(addr), "v"(pk) : "memory");
}

// ---------------------------------------------------------------------------
// Kernel 3: x f32 -> f16 (4 elements/thread).
// ---------------------------------------------------------------------------
__global__ __launch_bounds__(256) void ortho_cvt_kernel(
    const float* __restrict__ x, _Float16* __restrict__ xh, int n4) {
  const int i = blockIdx.x * blockDim.x + threadIdx.x;
  if (i >= n4) return;
  const float4 f = ((const float4*)x)[i];
  typedef __attribute__((ext_vector_type(4))) _Float16 v4h;
  v4h h;
  h[0] = (_Float16)f.x; h[1] = (_Float16)f.y;
  h[2] = (_Float16)f.z; h[3] = (_Float16)f.w;
  ((v4h*)xh)[i] = h;
}

// ---------------------------------------------------------------------------
// Kernel 4: WMMA GEMM. Workgroup = 256 threads = 8 waves computes a 256x128
// (MxO) output tile. Wave w owns a 32-row slab (two 16-row A fragments) x all
// 8 column tiles => 16 accumulator tiles; each staged B fragment feeds two
// WMMAs (load:wmma ratio 20:16 instead of 18:8).
// ---------------------------------------------------------------------------
__global__ __launch_bounds__(256) void ortho_gemm_kernel(
    const _Float16* __restrict__ xh,   // [M][I] row-major f16
    const _Float16* __restrict__ W,    // [O][I] row-major f16
    float* __restrict__ out,           // [M][O] f32
    int M) {
  __shared__ _Float16 Xs[256 * LDSTR];   // 20 KB
  __shared__ _Float16 Ws[128 * LDSTR];   // 10 KB

  const int tid   = threadIdx.x;
  const int wave  = tid >> 5;
  const int lane  = tid & 31;
  const int l16   = lane & 15;
  const int khalf = lane >> 4;           // 0 or 1
  const int mBase = blockIdx.y * 256;
  const int oBase = blockIdx.x * 128;

  v8f acc0[8] = {};   // rows wave*32 + [0,16)
  v8f acc1[8] = {};   // rows wave*32 + [16,32)

  for (int kb = 0; kb < I_FEAT; kb += 32) {
    // ---- cooperative stage.
    // X tile: 256 rows x 32 halves = 1024 16B chunks -> 4 per thread.
#pragma unroll
    for (int t = 0; t < 4; ++t) {
      const int c   = tid + t * 256;
      const int row = c >> 2;
      const int seg = c & 3;
      *(uint4*)(&Xs[row * LDSTR + seg * 8]) =
          *(const uint4*)(&xh[(size_t)(mBase + row) * I_FEAT + kb + seg * 8]);
    }
    // W tile: 128 rows x 32 halves = 512 16B chunks -> 2 per thread.
#pragma unroll
    for (int t = 0; t < 2; ++t) {
      const int c   = tid + t * 256;
      const int row = c >> 2;
      const int seg = c & 3;
      *(uint4*)(&Ws[row * LDSTR + seg * 8]) =
          *(const uint4*)(&W[(size_t)(oBase + row) * I_FEAT + kb + seg * 8]);
    }
    __syncthreads();

    // ---- two A fragments (16x32 f16): lane holds row m=l16; K chunks per
    // ISA layout: low-lane K {0..7,16..23}, high-lane K {8..15,24..31}.
    v16h a0, a1;
    {
      const _Float16* ar0 = &Xs[(wave * 32 + l16) * LDSTR + 8 * khalf];
      const _Float16* ar1 = &Xs[(wave * 32 + 16 + l16) * LDSTR + 8 * khalf];
      const v8h lo0 = *(const v8h*)(ar0);
      const v8h hi0 = *(const v8h*)(ar0 + 16);
      const v8h lo1 = *(const v8h*)(ar1);
      const v8h hi1 = *(const v8h*)(ar1 + 16);
#pragma unroll
      for (int i = 0; i < 8; ++i) {
        a0[i] = lo0[i]; a0[i + 8] = hi0[i];
        a1[i] = lo1[i]; a1[i + 8] = hi1[i];
      }
    }

    // ---- 8 shared B fragments (32x16 f16): lane holds column n=l16 (= a W
    // row), K chunks {8*khalf..+7, 16+8*khalf..+7}. Each feeds 2 WMMAs.
#pragma unroll
    for (int n = 0; n < 8; ++n) {
      const _Float16* br = &Ws[(n * 16 + l16) * LDSTR + 8 * khalf];
      const v8h lo = *(const v8h*)(br);
      const v8h hi = *(const v8h*)(br + 16);
      v16h b;
#pragma unroll
      for (int i = 0; i < 8; ++i) { b[i] = lo[i]; b[i + 8] = hi[i]; }
      acc0[n] = __builtin_amdgcn_wmma_f32_16x16x32_f16(
          false, a0, false, b, (short)0, acc0[n], false, false);
      acc1[n] = __builtin_amdgcn_wmma_f32_16x16x32_f16(
          false, a1, false, b, (short)0, acc1[n], false, false);
    }
    __syncthreads();
  }

  // ---- epilogue: C/D layout -> VGPR r holds M = r + 8*khalf, N = l16.
#pragma unroll
  for (int n = 0; n < 8; ++n) {
    const int o = oBase + n * 16 + l16;
#pragma unroll
    for (int r = 0; r < 8; ++r) {
      const int m0 = mBase + wave * 32 + r + 8 * khalf;
      const int m1 = m0 + 16;
      out[(size_t)m0 * O_FEAT + o] = acc0[n][r];
      out[(size_t)m1 * O_FEAT + o] = acc1[n][r];
    }
  }
}

// ---------------------------------------------------------------------------
extern "C" void kernel_launch(void* const* d_in, const int* in_sizes, int n_in,
                              void* d_out, int out_size, void* d_ws, size_t ws_size,
                              hipStream_t stream) {
  const float*         x           = (const float*)d_in[0];
  const unsigned char* base_packed = (const unsigned char*)d_in[1];
  const float*         scales      = (const float*)d_in[2];
  const _Float16*      ortho_vals  = (const _Float16*)d_in[3];
  const int*           ortho_rows  = (const int*)d_in[4];
  const int*           ortho_cols  = (const int*)d_in[5];
  const float*         alpha       = (const float*)d_in[6];

  const int M   = in_sizes[0] / I_FEAT;   // 2048
  const int nnz = in_sizes[3];

  _Float16* Wd = (_Float16*)d_ws;                                  // 32 MB
  _Float16* xh = (_Float16*)((char*)d_ws + (size_t)O_FEAT * I_FEAT * 2);

  // 1) dequantize int4 -> f16 W
  {
    const int threads = (O_FEAT * (I_FEAT / 2)) / 4;  // 2,097,152
    ortho_dequant_kernel<<<threads / 256, 256, 0, stream>>>(base_packed, scales, Wd);
  }
  // 2) scatter sparse residual (packed f16 global atomics)
  ortho_scatter_kernel<<<(nnz + 255) / 256, 256, 0, stream>>>(
      ortho_vals, ortho_rows, ortho_cols, alpha, Wd, nnz);
  // 3) x -> f16
  {
    const int n4 = (M * I_FEAT) / 4;
    ortho_cvt_kernel<<<(n4 + 255) / 256, 256, 0, stream>>>(x, xh, n4);
  }
  // 4) WMMA GEMM
  {
    dim3 grid(O_FEAT / 128, M / 256);
    ortho_gemm_kernel<<<grid, 256, 0, stream>>>(xh, Wd, (float*)d_out, M);
  }
}
